// MultiScaleTransformerConv_84207128805741
// MI455X (gfx1250) — matrix operand order repaired and checked by reading, not compiled, not run on hardware
//
#include <hip/hip_runtime.h>
#include <math.h>

#define NN 50000
#define EE 800000
#define DD 128      // K dimension of both GEMMs
#define PCOLS 512   // packed projection columns: [q_l k_l v_l s_l q_g k_g v_g s_g]
#define OCN 128

typedef __attribute__((ext_vector_type(16))) __bf16 v16bf;
typedef __attribute__((ext_vector_type(8)))  float  v8f;

union BFrag { v16bf v; unsigned int u32[8]; };
union AccF  { v8f   v; float        f[8];  };

__device__ __forceinline__ unsigned short f2bf(float f) {
  unsigned int u = __float_as_uint(f);
  unsigned int r = (u + 0x7FFFu + ((u >> 16) & 1u)) >> 16;   // round-to-nearest-even
  return (unsigned short)r;
}

__device__ __forceinline__ void atomicMaxFloat(float* addr, float val) {
  if (val >= 0.0f) atomicMax((int*)addr, __float_as_int(val));
  else             atomicMin((unsigned int*)addr, __float_as_uint(val));
}

// ---------------------------------------------------------------------------
// Pack fp32 weights -> bf16, transposed to [col][k] so B fragments are
// contiguous 32-bit LDS loads. Also gathers the 8 projection biases.
// ---------------------------------------------------------------------------
__global__ void pack_weights(
    const float* W0, const float* W1, const float* W2, const float* W3,
    const float* W4, const float* W5, const float* W6, const float* W7,
    const float* b0, const float* b1, const float* b2, const float* b3,
    const float* b4, const float* b5, const float* b6, const float* b7,
    const float* Wf,
    unsigned short* __restrict__ Wpk, float* __restrict__ b512,
    unsigned short* __restrict__ Wft)
{
  const float* Ws[8] = {W0, W1, W2, W3, W4, W5, W6, W7};
  const float* bs[8] = {b0, b1, b2, b3, b4, b5, b6, b7};
  int tid = blockIdx.x * blockDim.x + threadIdx.x;
  int stride = gridDim.x * blockDim.x;
  // Wpk[col][k]; source W is [k][c] (128x64), col = mat*64 + c
  for (int i = tid; i < 8 * 64 * 128; i += stride) {
    int k = i & 127, col = i >> 7;
    int mat = col >> 6, c = col & 63;
    Wpk[i] = f2bf(Ws[mat][k * 64 + c]);
  }
  for (int i = tid; i < 512; i += stride)
    b512[i] = bs[i >> 6][i & 63];
  // Wft[col][k]; Wf is [k][c] (128x128)
  for (int i = tid; i < 128 * 128; i += stride) {
    int k = i & 127, col = i >> 7;
    Wft[i] = f2bf(Wf[k * 128 + col]);
  }
}

__global__ void init_buffers(float* __restrict__ node_max,
                             float* __restrict__ node_sum,
                             float* __restrict__ node_out)
{
  int tid = blockIdx.x * blockDim.x + threadIdx.x;
  if (tid < (int)(NN * 128)) node_out[tid] = 0.0f;
  if (tid < NN * 2) {
    node_max[tid] = __uint_as_float(0xFF800000u);   // -inf
    node_sum[tid] = 0.0f;
  }
}

// ---------------------------------------------------------------------------
// C[M x ldc] = A[M x 128](fp32) @ Bt^T (Bt is bf16 [col][k]) + bias
// Block: 128 threads = 4 waves; tile BM=16, BN=64 (one 16x16 WMMA per wave
// per k-step). K=128 -> 4 x v_wmma_f32_16x16x32_bf16 per wave.
// ---------------------------------------------------------------------------
__global__ void __launch_bounds__(128)
gemm_k128_bf16(const float* __restrict__ A,
               const unsigned short* __restrict__ Bt,
               const float* __restrict__ bias,
               float* __restrict__ C, int ldc)
{
  __shared__ unsigned int sAu[16 * 128 / 2];   // 16 rows x 128 k, bf16
  __shared__ unsigned int sBu[64 * 128 / 2];   // 64 cols x 128 k, bf16
  unsigned short* sA = (unsigned short*)sAu;

  const int tid  = threadIdx.x;
  const int row0 = blockIdx.x * 16;
  const int col0 = blockIdx.y * 64;

  // stage A tile (fp32 -> bf16)
  for (int i = tid; i < 16 * 128; i += 128) {
    int r = i >> 7, k = i & 127;
    sA[i] = f2bf(A[(size_t)(row0 + r) * DD + k]);
  }
  // stage B slice (already bf16, [col][k] rows are 64 dwords each)
  const unsigned int* Bu = (const unsigned int*)Bt;
  for (int i = tid; i < 64 * 64; i += 128)
    sBu[i] = Bu[(size_t)col0 * 64 + i];
  __syncthreads();

  const int lane = tid & 31;
  const int wave = tid >> 5;
  const int n    = lane & 15;
  const int coll = wave * 16 + n;        // column within the 64-col slice
  const int hi   = (lane >> 4) & 1;      // lane group (ISA 16-bit layouts)
  const int m    = n;                    // A row handled by this lane

  AccF acc;
#pragma unroll
  for (int r = 0; r < 8; ++r) acc.f[r] = 0.0f;

#pragma unroll
  for (int kk = 0; kk < 128; kk += 32) {
    BFrag a, b;
#pragma unroll
    for (int i = 0; i < 8; ++i) {
      // A 16x32 bf16 layout: VGPR i holds K = 16*(i/4) + 8*hi + 2*(i%4) +{0,1}
      int ka = kk + ((i >> 2) << 4) + (hi << 3) + ((i & 3) << 1);
      a.u32[i] = *(const unsigned int*)(sA + m * 128 + ka);
      // B 32x16 bf16 layout: VGPR i holds K = 16*(i/4) + 8*hi + 2*(i%4) +{0,1}, col fixed
      int kb = kk + ((i >> 2) << 4) + (hi << 3) + ((i & 3) << 1);
      b.u32[i] = sBu[coll * 64 + (kb >> 1)];
    }
    acc.v = __builtin_amdgcn_wmma_f32_16x16x32_bf16(
        false, a.v, false, b.v, (short)0, acc.v, false, false);
  }

  const int col = col0 + coll;
  const float bcol = bias[col];
#pragma unroll
  for (int r = 0; r < 8; ++r) {
    int mm = r + (hi << 3);              // C/D: lanes 0-15 -> M=r, 16-31 -> M=r+8
    C[(size_t)(row0 + mm) * ldc + col] = acc.f[r] + bcol;
  }
}

// ---------------------------------------------------------------------------
// Edge pass 1: alpha = q[dst].(k[src]+ea*We)/8 for both convs -> atomic max
// One wave per edge; lane handles channels c and c+32 (coalesced).
// ---------------------------------------------------------------------------
__global__ void __launch_bounds__(256)
edge_pass1(const int* __restrict__ ei, const float* __restrict__ ea,
           const float* __restrict__ proj,
           const float* __restrict__ We_l, const float* __restrict__ We_g,
           float* __restrict__ node_max)
{
  int e = (blockIdx.x * blockDim.x + threadIdx.x) >> 5;
  if (e >= EE) return;
  int lane = threadIdx.x & 31;
  int src = ei[e], dst = ei[EE + e];
  float attr = ea[e];
  const float* q  = proj + (size_t)dst * PCOLS;
  const float* kv = proj + (size_t)src * PCOLS;
  int c0 = lane, c1 = lane + 32;
  float pl = q[c0] * (kv[64 + c0] + attr * We_l[c0])
           + q[c1] * (kv[64 + c1] + attr * We_l[c1]);
  float pg = q[256 + c0] * (kv[320 + c0] + attr * We_g[c0])
           + q[256 + c1] * (kv[320 + c1] + attr * We_g[c1]);
#pragma unroll
  for (int s = 16; s >= 1; s >>= 1) {
    pl += __shfl_xor(pl, s, 32);
    pg += __shfl_xor(pg, s, 32);
  }
  if (lane == 0) {
    atomicMaxFloat(&node_max[2 * dst],     pl * 0.125f);
    atomicMaxFloat(&node_max[2 * dst + 1], pg * 0.125f);
  }
}

// ---------------------------------------------------------------------------
// Edge pass 2: ex = exp(alpha - m[dst]); sum and v*ex scatter-adds (L2 atomics)
// ---------------------------------------------------------------------------
__global__ void __launch_bounds__(256)
edge_pass2(const int* __restrict__ ei, const float* __restrict__ ea,
           const float* __restrict__ proj,
           const float* __restrict__ We_l, const float* __restrict__ We_g,
           const float* __restrict__ node_max,
           float* __restrict__ node_sum, float* __restrict__ node_out)
{
  int e = (blockIdx.x * blockDim.x + threadIdx.x) >> 5;
  if (e >= EE) return;
  int lane = threadIdx.x & 31;
  int src = ei[e], dst = ei[EE + e];
  float attr = ea[e];
  const float* q  = proj + (size_t)dst * PCOLS;
  const float* kv = proj + (size_t)src * PCOLS;
  int c0 = lane, c1 = lane + 32;
  float pl = q[c0] * (kv[64 + c0] + attr * We_l[c0])
           + q[c1] * (kv[64 + c1] + attr * We_l[c1]);
  float pg = q[256 + c0] * (kv[320 + c0] + attr * We_g[c0])
           + q[256 + c1] * (kv[320 + c1] + attr * We_g[c1]);
#pragma unroll
  for (int s = 16; s >= 1; s >>= 1) {
    pl += __shfl_xor(pl, s, 32);
    pg += __shfl_xor(pg, s, 32);
  }
  float ml = node_max[2 * dst];     if (ml < -1e30f) ml = 0.0f;
  float mg = node_max[2 * dst + 1]; if (mg < -1e30f) mg = 0.0f;
  float exl = __expf(pl * 0.125f - ml);
  float exg = __expf(pg * 0.125f - mg);
  if (lane == 0) {
    atomicAdd(&node_sum[2 * dst],     exl);
    atomicAdd(&node_sum[2 * dst + 1], exg);
  }
  float* od = node_out + (size_t)dst * 128;
  atomicAdd(&od[c0],      (kv[128 + c0] + attr * We_l[c0]) * exl);
  atomicAdd(&od[c1],      (kv[128 + c1] + attr * We_l[c1]) * exl);
  atomicAdd(&od[64 + c0], (kv[384 + c0] + attr * We_g[c0]) * exg);
  atomicAdd(&od[64 + c1], (kv[384 + c1] + attr * We_g[c1]) * exg);
}

// ---------------------------------------------------------------------------
// Node finalize: normalize, beta-gated skip, write concat layout in place.
// One wave per node.
// ---------------------------------------------------------------------------
__global__ void __launch_bounds__(256)
node_finalize(const float* __restrict__ proj, const float* __restrict__ node_sum,
              const float* __restrict__ Wb_l, const float* __restrict__ Wb_g,
              float* __restrict__ node_out)
{
  int n = (blockIdx.x * blockDim.x + threadIdx.x) >> 5;
  if (n >= NN) return;
  int lane = threadIdx.x & 31;
  int c0 = lane, c1 = lane + 32;
  float invl = 1.0f / (node_sum[2 * n]     + 1e-16f);
  float invg = 1.0f / (node_sum[2 * n + 1] + 1e-16f);
  float* od = node_out + (size_t)n * 128;
  const float* pr = proj + (size_t)n * PCOLS;
  float ol0 = od[c0] * invl,      ol1 = od[c1] * invl;
  float og0 = od[64 + c0] * invg, og1 = od[64 + c1] * invg;
  float xl0 = pr[192 + c0], xl1 = pr[192 + c1];
  float xg0 = pr[448 + c0], xg1 = pr[448 + c1];
  float tl = ol0 * Wb_l[c0] + xl0 * Wb_l[64 + c0] + (ol0 - xl0) * Wb_l[128 + c0]
           + ol1 * Wb_l[c1] + xl1 * Wb_l[64 + c1] + (ol1 - xl1) * Wb_l[128 + c1];
  float tg = og0 * Wb_g[c0] + xg0 * Wb_g[64 + c0] + (og0 - xg0) * Wb_g[128 + c0]
           + og1 * Wb_g[c1] + xg1 * Wb_g[64 + c1] + (og1 - xg1) * Wb_g[128 + c1];
#pragma unroll
  for (int s = 16; s >= 1; s >>= 1) {
    tl += __shfl_xor(tl, s, 32);
    tg += __shfl_xor(tg, s, 32);
  }
  float bl = 1.0f / (1.0f + __expf(-tl));
  float bg = 1.0f / (1.0f + __expf(-tg));
  od[c0]      = bl * xl0 + (1.0f - bl) * ol0;
  od[c1]      = bl * xl1 + (1.0f - bl) * ol1;
  od[64 + c0] = bg * xg0 + (1.0f - bg) * og0;
  od[64 + c1] = bg * xg1 + (1.0f - bg) * og1;
}

extern "C" void kernel_launch(void* const* d_in, const int* in_sizes, int n_in,
                              void* d_out, int out_size, void* d_ws, size_t ws_size,
                              hipStream_t stream)
{
  const float* x   = (const float*)d_in[0];
  const int*   ei  = (const int*)  d_in[1];
  const float* ea  = (const float*)d_in[2];
  const float* lWq = (const float*)d_in[3];
  const float* lbq = (const float*)d_in[4];
  const float* lWk = (const float*)d_in[5];
  const float* lbk = (const float*)d_in[6];
  const float* lWv = (const float*)d_in[7];
  const float* lbv = (const float*)d_in[8];
  const float* lWe = (const float*)d_in[9];
  const float* lWs = (const float*)d_in[10];
  const float* lbs = (const float*)d_in[11];
  const float* lWb = (const float*)d_in[12];
  const float* gWq = (const float*)d_in[13];
  const float* gbq = (const float*)d_in[14];
  const float* gWk = (const float*)d_in[15];
  const float* gbk = (const float*)d_in[16];
  const float* gWv = (const float*)d_in[17];
  const float* gbv = (const float*)d_in[18];
  const float* gWe = (const float*)d_in[19];
  const float* gWs = (const float*)d_in[20];
  const float* gbs = (const float*)d_in[21];
  const float* gWb = (const float*)d_in[22];
  const float* Wf  = (const float*)d_in[23];
  const float* bf  = (const float*)d_in[24];

  // workspace layout (~129 MB, fits MI455X's 192 MB L2 working-set budget)
  char* ws = (char*)d_ws;
  size_t off = 0;
  float* proj = (float*)(ws + off);            off += (size_t)NN * PCOLS * 4;
  float* node_out = (float*)(ws + off);        off += (size_t)NN * 128 * 4;
  float* node_max = (float*)(ws + off);        off += (size_t)NN * 2 * 4;
  float* node_sum = (float*)(ws + off);        off += (size_t)NN * 2 * 4;
  unsigned short* Wpk = (unsigned short*)(ws + off); off += 512 * 128 * 2;
  float* b512 = (float*)(ws + off);            off += 512 * 4;
  unsigned short* Wft = (unsigned short*)(ws + off); off += 128 * 128 * 2;
  (void)ws_size; (void)in_sizes; (void)n_in; (void)out_size;

  pack_weights<<<128, 256, 0, stream>>>(
      lWq, lWk, lWv, lWs, gWq, gWk, gWv, gWs,
      lbq, lbk, lbv, lbs, gbq, gbk, gbv, gbs,
      Wf, Wpk, b512, Wft);

  init_buffers<<<(NN * 128 + 255) / 256, 256, 0, stream>>>(node_max, node_sum, node_out);

  dim3 g1(NN / 16, PCOLS / 64);                 // 3125 x 8
  gemm_k128_bf16<<<g1, 128, 0, stream>>>(x, Wpk, b512, proj, PCOLS);

  edge_pass1<<<(EE * 32 + 255) / 256, 256, 0, stream>>>(ei, ea, proj, lWe, gWe, node_max);
  edge_pass2<<<(EE * 32 + 255) / 256, 256, 0, stream>>>(ei, ea, proj, lWe, gWe,
                                                        node_max, node_sum, node_out);
  node_finalize<<<(NN * 32 + 255) / 256, 256, 0, stream>>>(proj, node_sum, lWb, gWb, node_out);

  dim3 g2(NN / 16, OCN / 64);                   // 3125 x 2
  gemm_k128_bf16<<<g2, 128, 0, stream>>>(node_out, Wft, bf, (float*)d_out, OCN);
}
